// TransformerEncoder_11424613007649
// MI455X (gfx1250) — 16-bit and 8-bit matrix operand order repaired and checked by reading, compiled, not run on hardware
//
#include <hip/hip_runtime.h>
#include <cstdint>
#include <cstddef>


typedef __attribute__((ext_vector_type(16))) _Float16     v16h;
typedef __attribute__((ext_vector_type(8)))  float        v8f;
typedef __attribute__((ext_vector_type(8)))  int          v8i;
typedef __attribute__((ext_vector_type(4)))  int          v4i;
typedef __attribute__((ext_vector_type(4)))  unsigned int v4u;

#define B_   4
#define D_   2048
#define S_   1024
#define H_   32
#define DH_  64
#define F_   8192
#define HD_  2048
#define KC_  256

#if defined(__has_builtin)
#if __has_builtin(__builtin_amdgcn_tensor_load_to_lds) && \
    __has_builtin(__builtin_amdgcn_s_wait_tensorcnt)
#define USE_TDM 1
#endif
#endif
#ifndef USE_TDM
#define USE_TDM 0
#endif

__device__ __forceinline__ int kA8(int v, int hi) {
  return ((v >> 1) << 4) + hi * 8 + ((v & 1) << 2);
}
__device__ __forceinline__ int kB8(int v, int hi) {
  return ((v >> 2) << 5) + hi * 16 + ((v & 3) << 2);
}
__device__ __forceinline__ int dA16(int j, int hi) {
  return ((j >> 3) << 4) + hi * 8 + (j & 7);
}

#if USE_TDM
__device__ __forceinline__ void tdm_stage_a(const int8_t* gTile, void* ldsDst,
                                            unsigned I, unsigned Orows) {
  const unsigned long long ga = (unsigned long long)(size_t)gTile;
  v4u g0;
  g0[0] = 1u;
  g0[1] = (unsigned)(size_t)ldsDst;
  g0[2] = (unsigned)(ga & 0xffffffffu);
  g0[3] = (unsigned)((ga >> 32) & 0x01ffffffu) | (2u << 30);
  const unsigned long long stride0 = I;
  v8i g1;
  g1[0] = 0;
  g1[1] = (int)((I & 0xffffu) << 16);
  g1[2] = (int)((I >> 16) | ((Orows & 0xffffu) << 16));
  g1[3] = (int)((Orows >> 16) | ((unsigned)KC_ << 16));
  g1[4] = (int)64;
  g1[5] = (int)(stride0 & 0xffffffffu);
  g1[6] = (int)((stride0 >> 32) & 0xffffu);
  g1[7] = 0;
  v4i gz = {0, 0, 0, 0};
#if __clang_major__ >= 23
  v8i gz8 = {0, 0, 0, 0, 0, 0, 0, 0};
  __builtin_amdgcn_tensor_load_to_lds(g0, g1, gz, gz, gz8, 0);
#else
  __builtin_amdgcn_tensor_load_to_lds(g0, g1, gz, gz, 0);
#endif
}
#endif

__global__ __launch_bounds__(256)
void wquant_kernel(const float* __restrict__ W, int8_t* __restrict__ qW,
                   float* __restrict__ sW, int I) {
  __shared__ float red[256];
  const int o = blockIdx.x, t = threadIdx.x;
  const float* row = W + (size_t)o * I;
  float amax = 0.f;
  for (int i = t; i < I; i += 256) amax = fmaxf(amax, fabsf(row[i]));
  red[t] = amax; __syncthreads();
  for (int w = 128; w > 0; w >>= 1) {
    if (t < w) red[t] = fmaxf(red[t], red[t + w]);
    __syncthreads();
  }
  const float scale = fmaxf(red[0] * (1.f / 127.f), 1e-8f);
  if (t == 0) sW[o] = scale;
  const float inv = 1.f / scale;
  int8_t* qrow = qW + (size_t)o * I;
  for (int i = t; i < I; i += 256) {
    float q = fminf(fmaxf(rintf(row[i] * inv), -127.f), 127.f);
    qrow[i] = (int8_t)q;
  }
}

__global__ __launch_bounds__(256)
void colquant_kernel(const float* __restrict__ X, const float* __restrict__ lnw,
                     int use_rms, int8_t* __restrict__ qT,
                     float* __restrict__ sX, int Din) {
  __shared__ float red[256];
  __shared__ float bc;
  const int s = blockIdx.x, b = blockIdx.y, t = threadIdx.x;
  const float* col = X + (size_t)b * Din * S_ + s;
  float r = 1.f;
  if (use_rms) {
    float ss = 0.f;
    for (int d = t; d < Din; d += 256) {
      float v = col[(size_t)d * S_];
      ss += v * v;
    }
    red[t] = ss; __syncthreads();
    for (int w = 128; w > 0; w >>= 1) {
      if (t < w) red[t] += red[t + w];
      __syncthreads();
    }
    if (t == 0) bc = rsqrtf(red[0] / (float)Din + 1e-5f);
    __syncthreads();
    r = bc;
    __syncthreads();
  }
  float amax = 0.f;
  for (int d = t; d < Din; d += 256) {
    float v = col[(size_t)d * S_] * r * (use_rms ? lnw[d] : 1.f);
    amax = fmaxf(amax, fabsf(v));
  }
  red[t] = amax; __syncthreads();
  for (int w = 128; w > 0; w >>= 1) {
    if (t < w) red[t] = fmaxf(red[t], red[t + w]);
    __syncthreads();
  }
  const float scale = fmaxf(red[0] * (1.f / 127.f), 1e-8f);
  if (t == 0) sX[(size_t)b * S_ + s] = scale;
  const float inv = 1.f / scale;
  int8_t* q = qT + ((size_t)b * S_ + s) * Din;
  for (int d = t; d < Din; d += 256) {
    float v = col[(size_t)d * S_] * r * (use_rms ? lnw[d] : 1.f);
    q[d] = (int8_t)fminf(fmaxf(rintf(v * inv), -127.f), 127.f);
  }
}

__global__ __launch_bounds__(128)
void qgemm_kernel(const int8_t* __restrict__ qW, const float* __restrict__ sW,
                  const int8_t* __restrict__ qXT, const float* __restrict__ sX,
                  void* __restrict__ out, const float* __restrict__ residual,
                  int O, int I, int mode) {
  __shared__ int8_t ldsA[2][64 * KC_];
  const int lane = threadIdx.x & 31;
  const int wave = threadIdx.x >> 5;
  const int m = lane & 15, hi = lane >> 4;
  const int oBase = blockIdx.y * 64;
  const int sBase = blockIdx.x * 64;
  const int b = blockIdx.z;
  const int nChunk = I / KC_;

  const int8_t* gA = qW + (size_t)oBase * I;

  const int8_t* colB[4];
#pragma unroll
  for (int sub = 0; sub < 4; ++sub)
    colB[sub] = qXT + ((size_t)b * S_ + sBase + sub * 16 + m) * I;

  union I8U { v8i v; int e[8]; };
  I8U acc[4];
#pragma unroll
  for (int sub = 0; sub < 4; ++sub)
#pragma unroll
    for (int e = 0; e < 8; ++e) acc[sub].e[e] = 0;

#if USE_TDM
  if (wave == 0) tdm_stage_a(gA, &ldsA[0][0], (unsigned)I, (unsigned)O);
#else
  for (int idx = threadIdx.x; idx < 64 * (KC_ / 16); idx += 128) {
    const int row = idx >> 4, off = (idx & 15) * 16;
    *reinterpret_cast<uint4*>(&ldsA[0][row * KC_ + off]) =
        *reinterpret_cast<const uint4*>(gA + (size_t)row * I + off);
  }
#endif

#pragma unroll 1
  for (int c = 0; c < nChunk; ++c) {
    const int buf = c & 1;
#if USE_TDM
    if (wave == 0) {
      if (c + 1 < nChunk) {
        tdm_stage_a(gA + (size_t)(c + 1) * KC_, &ldsA[buf ^ 1][0],
                    (unsigned)I, (unsigned)O);
        __builtin_amdgcn_s_wait_tensorcnt((short)1);
      } else {
        __builtin_amdgcn_s_wait_tensorcnt((short)0);
      }
    }
#else
    if (c + 1 < nChunk) {
      const int8_t* src = gA + (size_t)(c + 1) * KC_;
      for (int idx = threadIdx.x; idx < 64 * (KC_ / 16); idx += 128) {
        const int row = idx >> 4, off = (idx & 15) * 16;
        *reinterpret_cast<uint4*>(&ldsA[buf ^ 1][row * KC_ + off]) =
            *reinterpret_cast<const uint4*>(src + (size_t)row * I + off);
      }
    }
#endif
    __syncthreads();

#pragma unroll
    for (int kk = 0; kk < 4; ++kk) {
      I8U a;
      const int8_t* aRow = &ldsA[buf][(wave * 16 + m) * KC_ + kk * 64];
#pragma unroll
      for (int v = 0; v < 8; ++v)
        a.e[v] = *reinterpret_cast<const int*>(aRow + kA8(v, hi));
      const int k0 = c * KC_ + kk * 64;
#pragma unroll
      for (int sub = 0; sub < 4; ++sub) {
        if (kk == 0) __builtin_prefetch(colB[sub] + k0 + KC_, 0, 0);
        I8U bb;
#pragma unroll
        for (int v = 0; v < 8; ++v)
          bb.e[v] = *reinterpret_cast<const int*>(colB[sub] + k0 + kA8(v, hi));
        acc[sub].v = __builtin_amdgcn_wmma_i32_16x16x64_iu8(
             true, a.v,  true, bb.v, acc[sub].v,
             false,  false);
      }
    }
    __syncthreads();
  }

#pragma unroll
  for (int sub = 0; sub < 4; ++sub) {
    const int sc = sBase + sub * 16 + m;
    const float sxc = sX[(size_t)b * S_ + sc];
#pragma unroll
    for (int r = 0; r < 8; ++r) {
      const int o = oBase + wave * 16 + r + hi * 8;
      const float f = (float)acc[sub].e[r] * sW[o] * sxc;
      if (mode == 0) {
        ((_Float16*)out)[(size_t)b * O * S_ + (size_t)o * S_ + sc] = (_Float16)f;
      } else if (mode == 1) {
        const int h = o >> 6, d = o & 63;
        ((_Float16*)out)[(((size_t)b * H_ + h) * S_ + sc) * DH_ + d] = (_Float16)f;
      } else {
        const size_t idx = (size_t)b * O * S_ + (size_t)o * S_ + sc;
        ((float*)out)[idx] = f + residual[idx];
      }
    }
  }
}

__global__ __launch_bounds__(32)
void attention_kernel(const _Float16* __restrict__ Q,
                      const _Float16* __restrict__ KT,
                      const _Float16* __restrict__ V,
                      const float* __restrict__ bias,
                      const int* __restrict__ mask,
                      float* __restrict__ attOut) {
  __shared__ __align__(32) _Float16 Plds[16 * 32];
  const int lane = threadIdx.x;
  const int n = lane & 15, hi = lane >> 4;
  const int qt = blockIdx.x, h = blockIdx.y, b = blockIdx.z;
  const int qg = qt * 16 + n;

  const _Float16* Qb = Q + ((size_t)b * HD_ + (size_t)h * DH_) * S_;
  const _Float16* Vb = V + ((size_t)b * HD_ + (size_t)h * DH_) * S_;
  const _Float16* Kb = KT + ((size_t)b * H_ + h) * (size_t)S_ * DH_;
  const float* biasb = bias + (size_t)h * S_ * S_;
  const int* maskb = mask + (size_t)b * S_ * S_;

  union HF  { v16h v; _Float16 x[16]; };
  union F8U { v8f v; float f[8]; };

  HF bq0, bq1;
#pragma unroll
  for (int j = 0; j < 16; ++j) {
    const int d = dA16(j, hi);
    bq0.x[j] = Qb[(size_t)d * S_ + qg];
    bq1.x[j] = Qb[(size_t)(d + 32) * S_ + qg];
  }

  F8U acc[4];
  for (int dt = 0; dt < 4; ++dt)
    for (int e = 0; e < 8; ++e) acc[dt].f[e] = 0.f;
  float mrow = -1e30f, lrow = 0.f;

  for (int k0 = 0; k0 < S_; k0 += 32) {
    F8U sc[2];
    float pmax = -1e30f;
#pragma unroll
    for (int sub = 0; sub < 2; ++sub) {
      HF aK0, aK1;
      const _Float16* krp = Kb + (size_t)(k0 + sub * 16 + n) * DH_;
#pragma unroll
      for (int j = 0; j < 16; ++j) {
        const int d = dA16(j, hi);
        aK0.x[j] = krp[d];
        aK1.x[j] = krp[d + 32];
      }
      v8f cz = {};
      v8f c = __builtin_amdgcn_wmma_f32_16x16x32_f16(
          false, aK0.v, false, bq0.v, (short)0, cz, false, false);
      c = __builtin_amdgcn_wmma_f32_16x16x32_f16(
          false, aK1.v, false, bq1.v, (short)0, c, false, false);
      sc[sub].v = c;
#pragma unroll
      for (int r = 0; r < 8; ++r) {
        const int kg = k0 + sub * 16 + r + hi * 8;
        float sv = sc[sub].f[r] + biasb[(size_t)kg * S_ + qg];
        sv = maskb[(size_t)kg * S_ + qg] ? sv : -1e30f;
        sc[sub].f[r] = sv;
        pmax = fmaxf(pmax, sv);
      }
    }
    pmax = fmaxf(pmax, __shfl_xor(pmax, 16, 32));
    const float mnew = fmaxf(mrow, pmax);
    const float alpha = __expf(mrow - mnew);
    float psum = 0.f;
#pragma unroll
    for (int sub = 0; sub < 2; ++sub)
#pragma unroll
      for (int r = 0; r < 8; ++r) {
        const float p = __expf(sc[sub].f[r] - mnew);
        sc[sub].f[r] = p;
        psum += p;
      }
    psum += __shfl_xor(psum, 16, 32);
    lrow = lrow * alpha + psum;
    mrow = mnew;
    for (int dt = 0; dt < 4; ++dt)
#pragma unroll
      for (int e = 0; e < 8; ++e) acc[dt].f[e] *= alpha;

#pragma unroll
    for (int sub = 0; sub < 2; ++sub)
#pragma unroll
      for (int r = 0; r < 8; ++r)
        Plds[n * 32 + sub * 16 + r + hi * 8] = (_Float16)sc[sub].f[r];
    __syncthreads();
    HF bp;
    bp.v = *reinterpret_cast<const v16h*>(Plds + n * 32 + hi * 16);

#pragma unroll
    for (int dt = 0; dt < 4; ++dt) {
      HF aV;
      const _Float16* vrp = Vb + (size_t)(dt * 16 + n) * S_ + k0;
#pragma unroll
      for (int j = 0; j < 16; ++j) aV.x[j] = vrp[hi * 16 + j];
      acc[dt].v = __builtin_amdgcn_wmma_f32_16x16x32_f16(
          false, aV.v, false, bp.v, (short)0, acc[dt].v, false, false);
    }
    __syncthreads();
  }

  const float rl = 1.f / lrow;
  for (int dt = 0; dt < 4; ++dt)
#pragma unroll
    for (int r = 0; r < 8; ++r) {
      const int d = dt * 16 + r + hi * 8;
      attOut[(size_t)b * HD_ * S_ + ((size_t)h * DH_ + d) * S_ + qg] =
          acc[dt].f[r] * rl;
    }
}

__device__ __forceinline__ float gelu_tanh(float x) {
  const float c = 0.7978845608028654f;
  return 0.5f * x * (1.f + tanhf(c * (x + 0.044715f * x * x * x)));
}

__global__ __launch_bounds__(256)
void geglu_quant_kernel(const _Float16* __restrict__ ff0,
                        const _Float16* __restrict__ ff1,
                        int8_t* __restrict__ gT, float* __restrict__ sG) {
  __shared__ float red[256];
  const int s = blockIdx.x, b = blockIdx.y, t = threadIdx.x;
  const size_t base = (size_t)b * F_ * S_ + s;
  float amax = 0.f;
  for (int f = t; f < F_; f += 256) {
    const float g = gelu_tanh((float)ff0[base + (size_t)f * S_]) *
                    (float)ff1[base + (size_t)f * S_];
    amax = fmaxf(amax, fabsf(g));
  }
  red[t] = amax; __syncthreads();
  for (int w = 128; w > 0; w >>= 1) {
    if (t < w) red[t] = fmaxf(red[t], red[t + w]);
    __syncthreads();
  }
  const float scale = fmaxf(red[0] * (1.f / 127.f), 1e-8f);
  if (t == 0) sG[(size_t)b * S_ + s] = scale;
  const float inv = 1.f / scale;
  int8_t* q = gT + ((size_t)b * S_ + s) * F_;
  for (int f = t; f < F_; f += 256) {
    const float g = gelu_tanh((float)ff0[base + (size_t)f * S_]) *
                    (float)ff1[base + (size_t)f * S_];
    q[f] = (int8_t)fminf(fmaxf(rintf(g * inv), -127.f), 127.f);
  }
}

extern "C" void kernel_launch(void* const* d_in, const int* in_sizes, int n_in,
                              void* d_out, int out_size, void* d_ws, size_t ws_size,
                              hipStream_t stream) {
  (void)in_sizes; (void)n_in; (void)out_size; (void)ws_size;

  const float* hidden = (const float*)d_in[0];
  const int*   mask   = (const int*)d_in[1];
  const float* pbias  = (const float*)d_in[2];
  const float* ln1    = (const float*)d_in[3];
  const float* ln2    = (const float*)d_in[4];
  const float* Wq     = (const float*)d_in[5];
  const float* Wk     = (const float*)d_in[6];
  const float* Wv     = (const float*)d_in[7];
  const float* Wo     = (const float*)d_in[8];
  const float* w0     = (const float*)d_in[9];
  const float* w1     = (const float*)d_in[10];
  const float* w_out  = (const float*)d_in[11];
  float* outp = (float*)d_out;

  char* ws = (char*)d_ws;
  size_t off = 0;
  auto carve = [&](size_t bytes) -> void* {
    off = (off + 255) & ~(size_t)255;
    void* p = ws + off;
    off += bytes;
    return p;
  };

  int8_t* qWq   = (int8_t*)carve((size_t)HD_ * D_);
  int8_t* qWk   = (int8_t*)carve((size_t)HD_ * D_);
  int8_t* qWv   = (int8_t*)carve((size_t)HD_ * D_);
  int8_t* qWo   = (int8_t*)carve((size_t)D_ * HD_);
  int8_t* qw0   = (int8_t*)carve((size_t)F_ * D_);
  int8_t* qw1   = (int8_t*)carve((size_t)F_ * D_);
  int8_t* qwout = (int8_t*)carve((size_t)D_ * F_);
  float* sWq   = (float*)carve(HD_ * 4);
  float* sWk   = (float*)carve(HD_ * 4);
  float* sWv   = (float*)carve(HD_ * 4);
  float* sWo   = (float*)carve(D_ * 4);
  float* sw0   = (float*)carve(F_ * 4);
  float* sw1   = (float*)carve(F_ * 4);
  float* swout = (float*)carve(D_ * 4);

  int8_t* qXT  = (int8_t*)carve((size_t)B_ * S_ * D_);
  float*  sX   = (float*)carve((size_t)B_ * S_ * 4);
  _Float16* Qf16  = (_Float16*)carve((size_t)B_ * HD_ * S_ * 2);
  _Float16* KTf16 = (_Float16*)carve((size_t)B_ * HD_ * S_ * 2);
  _Float16* Vf16  = (_Float16*)carve((size_t)B_ * HD_ * S_ * 2);
  float*  attbuf = (float*)carve((size_t)B_ * HD_ * S_ * 4);
  int8_t* attT   = (int8_t*)carve((size_t)B_ * S_ * HD_);
  float*  sAtt   = (float*)carve((size_t)B_ * S_ * 4);
  float*  hid    = (float*)carve((size_t)B_ * D_ * S_ * 4);
  int8_t* qYT    = (int8_t*)carve((size_t)B_ * S_ * D_);
  float*  sY     = (float*)carve((size_t)B_ * S_ * 4);
  _Float16* ff0  = (_Float16*)carve((size_t)B_ * F_ * S_ * 2);
  _Float16* ff1  = (_Float16*)carve((size_t)B_ * F_ * S_ * 2);
  int8_t* gT     = (int8_t*)carve((size_t)B_ * S_ * F_);
  float*  sG     = (float*)carve((size_t)B_ * S_ * 4);

  wquant_kernel<<<HD_, 256, 0, stream>>>(Wq, qWq, sWq, D_);
  wquant_kernel<<<HD_, 256, 0, stream>>>(Wk, qWk, sWk, D_);
  wquant_kernel<<<HD_, 256, 0, stream>>>(Wv, qWv, sWv, D_);
  wquant_kernel<<<D_,  256, 0, stream>>>(Wo, qWo, sWo, HD_);
  wquant_kernel<<<F_,  256, 0, stream>>>(w0, qw0, sw0, D_);
  wquant_kernel<<<F_,  256, 0, stream>>>(w1, qw1, sw1, D_);
  wquant_kernel<<<D_,  256, 0, stream>>>(w_out, qwout, swout, F_);

  colquant_kernel<<<dim3(S_, B_), 256, 0, stream>>>(hidden, ln1, 1, qXT, sX, D_);

  dim3 gqkv(S_ / 64, HD_ / 64, B_);
  qgemm_kernel<<<gqkv, 128, 0, stream>>>(qWq, sWq, qXT, sX, Qf16, nullptr, HD_, D_, 0);
  qgemm_kernel<<<gqkv, 128, 0, stream>>>(qWk, sWk, qXT, sX, KTf16, nullptr, HD_, D_, 1);
  qgemm_kernel<<<gqkv, 128, 0, stream>>>(qWv, sWv, qXT, sX, Vf16, nullptr, HD_, D_, 0);

  attention_kernel<<<dim3(S_ / 16, H_, B_), 32, 0, stream>>>(
      Qf16, KTf16, Vf16, pbias, mask, attbuf);

  colquant_kernel<<<dim3(S_, B_), 256, 0, stream>>>(attbuf, nullptr, 0, attT, sAtt, HD_);
  qgemm_kernel<<<dim3(S_ / 64, D_ / 64, B_), 128, 0, stream>>>(
      qWo, sWo, attT, sAtt, hid, hidden, D_, HD_, 2);

  colquant_kernel<<<dim3(S_, B_), 256, 0, stream>>>(hid, ln2, 1, qYT, sY, D_);
  dim3 gffn(S_ / 64, F_ / 64, B_);
  qgemm_kernel<<<gffn, 128, 0, stream>>>(qw0, sw0, qYT, sY, ff0, nullptr, F_, D_, 0);
  qgemm_kernel<<<gffn, 128, 0, stream>>>(qw1, sw1, qYT, sY, ff1, nullptr, F_, D_, 0);

  geglu_quant_kernel<<<dim3(S_, B_), 256, 0, stream>>>(ff0, ff1, gT, sG);
  qgemm_kernel<<<dim3(S_ / 64, D_ / 64, B_), 128, 0, stream>>>(
      qwout, swout, gT, sG, outp, hid, D_, F_, 2);
}
